// MultiMessagePassingWithGlobalNode_4870492914029
// MI455X (gfx1250) — matrix operand order repaired and checked by reading, not compiled, not run on hardware
//
#include <hip/hip_runtime.h>

#define EMB   128
#define EDGEF 32
#define NSTEPS 3
#define GNUM  256

typedef __bf16 bf16_t;
typedef __attribute__((ext_vector_type(16))) __bf16 v16bf;
typedef __attribute__((ext_vector_type(8)))  float  v8f;

// RNE conversion (used only in the tiny weight-repack kernel)
__device__ __forceinline__ bf16_t f2bf(float f) {
  unsigned u = __builtin_bit_cast(unsigned, f);
  unsigned r = u + 0x7FFFu + ((u >> 16) & 1u);
  unsigned short h = (unsigned short)(r >> 16);
  return __builtin_bit_cast(bf16_t, h);
}

// Fast pair conversion for activations: round-half-up + v_perm_b32 byte pack.
// dst = { bf16(b) , bf16(a) }  (a in low 16 bits)
__device__ __forceinline__ unsigned pack2(float a, float b) {
  unsigned ua = __builtin_bit_cast(unsigned, a) + 0x8000u;
  unsigned ub = __builtin_bit_cast(unsigned, b) + 0x8000u;
  return __builtin_amdgcn_perm(ub, ua, 0x07060302u);
}

__device__ __forceinline__ float lrelu(float v) { return v > 0.f ? v : 0.01f * v; }
__device__ __forceinline__ float zneg(float f)  { return f < -3.0e38f ? 0.f : f; } // -inf -> 0 (PyG empty-segment fill)

__device__ __forceinline__ v8f wmma_bf16(v16bf a, v16bf b, v8f c) {
  // D = A(16x32 bf16) * B(32x16 bf16) + C(16x16 f32)
  return __builtin_amdgcn_wmma_f32_16x16x32_bf16(false, a, false, b, (short)0, c, false, false);
}

// A-fragment: lane<16 holds row (lane&15), K = {p0[0..7], p1[0..7]} per ISA 16-bit A layout
__device__ __forceinline__ v16bf make_afrag(const float* p0, const float* p1) {
  union { v16bf v; unsigned u[8]; } f;
#pragma unroll
  for (int i = 0; i < 4; ++i) f.u[i]     = pack2(p0[2 * i], p0[2 * i + 1]);
#pragma unroll
  for (int i = 0; i < 4; ++i) f.u[4 + i] = pack2(p1[2 * i], p1[2 * i + 1]);
  return f.v;
}

// float atomic max via monotone int/uint bit-pattern trick (init location to -inf bits)
__device__ __forceinline__ void atomicMaxF(float* addr, float v) {
  if (v >= 0.f) atomicMax((int*)addr, __float_as_int(v));
  else          atomicMin((unsigned int*)addr, __float_as_uint(v));
}

__global__ void fill_u32(unsigned* __restrict__ p, unsigned v, long long n) {
  long long i = (long long)blockIdx.x * blockDim.x + threadIdx.x;
  if (i < n) p[i] = v;
}

// Repack W[K,128] row-major (f32) into WMMA B fragments [kt][nt][lane][16] (bf16):
// lane<16 -> column nt*16+(lane&15), K = kt*32 + {0..7,16..23} ; lane>=16 -> same col, K = kt*32 + {8..15,24..31}
__global__ void pack_b_frags(const float* __restrict__ W, bf16_t* __restrict__ out, int K) {
  int id = blockIdx.x * 256 + threadIdx.x;
  if (id >= K * EMB) return;
  int j    = id & 15;
  int lane = (id >> 4) & 31;
  int f    = id >> 9;
  int kt   = f >> 3, nt = f & 7;
  int n    = nt * 16 + (lane & 15);
  int k    = kt * 32 + ((j >> 3) << 4) + ((lane >> 4) << 3) + (j & 7);
  out[id] = f2bf(W[k * EMB + n]);
}

// ---- Fused edge message GEMM + scatter-max -------------------------------
// One wave = 32 edges (two M=16 tiles sharing every B fragment).
// K = 160 = [x[src] (128) | edge_attr (32)]. B fragments staged in LDS (40KB).
__global__ __launch_bounds__(256) void edge_msg_scattermax(
    const float* __restrict__ x, const float* __restrict__ ea,
    const long long* __restrict__ eidx,
    const bf16_t* __restrict__ Wfrag, const float* __restrict__ bm,
    float* __restrict__ agg, int E)
{
  __shared__ uint4 ldsW4[2560];                 // 40 KB: 5 kt * 8 nt * 32 lanes * 32B
  const uint4* g4 = (const uint4*)Wfrag;
  for (int i = threadIdx.x; i < 2560; i += 256) ldsW4[i] = g4[i];
  __syncthreads();
  const bf16_t* ldsW = (const bf16_t*)ldsW4;

  int lane = threadIdx.x & 31;
  int wave = threadIdx.x >> 5;
  int base = (blockIdx.x * 8 + wave) * 32;
  if (base >= E) return;                        // wave-uniform: EXEC stays all-ones for WMMA
  int half = lane >> 4;
  int er0 = base + (lane & 15);       if (er0 >= E) er0 = E - 1;
  int er1 = base + 16 + (lane & 15);  if (er1 >= E) er1 = E - 1;
  const long long* srcI = eidx;
  const long long* dstI = eidx + E;
  const float* xr0 = x + (long long)((int)srcI[er0]) * EMB;
  const float* xr1 = x + (long long)((int)srcI[er1]) * EMB;
  const float* ar0 = ea + (long long)er0 * EDGEF;
  const float* ar1 = ea + (long long)er1 * EDGEF;

  v8f acc0[8] = {};
  v8f acc1[8] = {};
#pragma unroll
  for (int kt = 0; kt < 5; ++kt) {
    const float* r0 = (kt < 4) ? (xr0 + kt * 32) : ar0;
    const float* r1 = (kt < 4) ? (xr1 + kt * 32) : ar1;
    v16bf a0 = make_afrag(r0 + half * 8, r0 + 16 + half * 8);
    v16bf a1 = make_afrag(r1 + half * 8, r1 + 16 + half * 8);
#pragma unroll
    for (int nt = 0; nt < 8; ++nt) {
      v16bf b = *(const v16bf*)(ldsW + (((kt * 8 + nt) * 32 + lane) << 4));
      acc0[nt] = wmma_bf16(a0, b, acc0[nt]);
      acc1[nt] = wmma_bf16(a1, b, acc1[nt]);
    }
  }
#pragma unroll
  for (int t = 0; t < 2; ++t) {
    int rbase = base + t * 16;
#pragma unroll
    for (int nt = 0; nt < 8; ++nt) {
      int col = nt * 16 + (lane & 15);
      float bias = bm[col];
#pragma unroll
      for (int i = 0; i < 8; ++i) {
        int eo = rbase + i + half * 8;         // D layout: M = vgpr + 8*(lane>=16)
        if (eo < E) {
          float v = lrelu((t ? acc1[nt][i] : acc0[nt][i]) + bias);
          int d = (int)dstI[eo];
          atomicMaxF(agg + (long long)d * EMB + col, v);  // L2-resident segment-max
        }
      }
    }
  }
}

// ---- Node update: [x | xg[batch] | agg] (K=384) @ Wa + residual ----------
__global__ __launch_bounds__(256) void node_update(
    const float* __restrict__ x, const float* __restrict__ xg,
    const float* __restrict__ agg, const long long* __restrict__ batch,
    const bf16_t* __restrict__ Wfrag, const float* __restrict__ ba,
    float* __restrict__ xout, int N)
{
  int lane = threadIdx.x & 31;
  int wave = threadIdx.x >> 5;
  int base = (blockIdx.x * 8 + wave) * 16;
  if (base >= N) return;
  int half = lane >> 4;
  int node = base + (lane & 15);
  if (node >= N) node = N - 1;
  const float* xr = x   + (long long)node * EMB;
  const float* gr = xg  + (long long)batch[node] * EMB;
  const float* ar = agg + (long long)node * EMB;

  v8f acc[8] = {};
#pragma unroll
  for (int kt = 0; kt < 12; ++kt) {
    v16bf a;
    if (kt < 8) {
      const float* row = (kt < 4) ? (xr + kt * 32) : (gr + (kt - 4) * 32);
      a = make_afrag(row + half * 8, row + 16 + half * 8);
    } else {
      const float* row = ar + (kt - 8) * 32;
      float t0[8], t1[8];
#pragma unroll
      for (int i = 0; i < 8; ++i) { t0[i] = zneg(row[half * 8 + i]); t1[i] = zneg(row[16 + half * 8 + i]); }
      a = make_afrag(t0, t1);
    }
#pragma unroll
    for (int nt = 0; nt < 8; ++nt) {
      v16bf b = *(const v16bf*)(Wfrag + (((kt * 8 + nt) * 32 + lane) << 4));
      acc[nt] = wmma_bf16(a, b, acc[nt]);
    }
  }
#pragma unroll
  for (int nt = 0; nt < 8; ++nt) {
    int col = nt * 16 + (lane & 15);
    float bias = ba[col];
#pragma unroll
    for (int i = 0; i < 8; ++i) {
      int no = base + i + half * 8;
      if (no < N) {
        long long off = (long long)no * EMB + col;
        xout[off] = lrelu(acc[nt][i] + bias) + x[off];
      }
    }
  }
}

// ---- gate = x@Wgate + b; segment max ------------------------------------
__global__ void gate_max(const float* __restrict__ x, const float* __restrict__ wg,
                         const float* __restrict__ bg, const long long* __restrict__ batch,
                         float* __restrict__ gate, float* __restrict__ gmax, int N)
{
  int n = blockIdx.x * 256 + threadIdx.x;
  if (n >= N) return;
  const float4* xr = (const float4*)(x + (long long)n * EMB);
  const float4* w  = (const float4*)wg;
  float acc = 0.f;
#pragma unroll
  for (int i = 0; i < 32; ++i) {
    float4 a = xr[i], b = w[i];
    acc += a.x * b.x + a.y * b.y + a.z * b.z + a.w * b.w;
  }
  acc += bg[0];
  gate[n] = acc;
  atomicMaxF(gmax + (int)batch[n], acc);
}

__global__ void softmax_den(const float* __restrict__ gate, const float* __restrict__ gmax,
                            const long long* __restrict__ batch, float* __restrict__ evec,
                            float* __restrict__ denom, int N)
{
  int n = blockIdx.x * 256 + threadIdx.x;
  if (n >= N) return;
  int b = (int)batch[n];
  float ev = __expf(gate[n] - gmax[b]);
  evec[n] = ev;
  atomicAdd(denom + b, ev);
}

// ---- feat = lrelu(x@Wfeat+b); pooled[b] += alpha * feat ------------------
__global__ __launch_bounds__(256) void feat_pool(
    const float* __restrict__ x, const long long* __restrict__ batch,
    const float* __restrict__ evec, const float* __restrict__ denom,
    const bf16_t* __restrict__ Wfrag, const float* __restrict__ bfe,
    float* __restrict__ pooled, int N)
{
  int lane = threadIdx.x & 31;
  int wave = threadIdx.x >> 5;
  int base = (blockIdx.x * 8 + wave) * 16;
  if (base >= N) return;
  int half = lane >> 4;
  int node = base + (lane & 15);
  if (node >= N) node = N - 1;
  const float* xr = x + (long long)node * EMB;

  v8f acc[8] = {};
#pragma unroll
  for (int kt = 0; kt < 4; ++kt) {
    const float* row = xr + kt * 32;
    v16bf a = make_afrag(row + half * 8, row + 16 + half * 8);
#pragma unroll
    for (int nt = 0; nt < 8; ++nt) {
      v16bf b = *(const v16bf*)(Wfrag + (((kt * 8 + nt) * 32 + lane) << 4));
      acc[nt] = wmma_bf16(a, b, acc[nt]);
    }
  }
  float alpha[8]; int bb[8];
#pragma unroll
  for (int i = 0; i < 8; ++i) {
    int no = base + i + half * 8;
    if (no >= N) no = N - 1;
    bb[i] = (int)batch[no];
    alpha[i] = evec[no] / denom[bb[i]];
  }
#pragma unroll
  for (int nt = 0; nt < 8; ++nt) {
    int col = nt * 16 + (lane & 15);
    float bias = bfe[col];
#pragma unroll
    for (int i = 0; i < 8; ++i) {
      if (base + i + half * 8 < N) {
        float f = lrelu(acc[nt][i] + bias);
        atomicAdd(pooled + (long long)bb[i] * EMB + col, alpha[i] * f);
      }
    }
  }
}

// ---- xg update: [pooled | xg] (K=256) @ Wt + residual --------------------
__global__ __launch_bounds__(256) void global_update(
    const float* __restrict__ pooled, const float* __restrict__ xg,
    const bf16_t* __restrict__ Wfrag, const float* __restrict__ bt,
    float* __restrict__ xgout)
{
  int lane = threadIdx.x & 31;
  int wave = threadIdx.x >> 5;
  int base = (blockIdx.x * 8 + wave) * 16;
  if (base >= GNUM) return;
  int half = lane >> 4;
  int g = base + (lane & 15);
  const float* pr = pooled + (long long)g * EMB;
  const float* gr = xg     + (long long)g * EMB;

  v8f acc[8] = {};
#pragma unroll
  for (int kt = 0; kt < 8; ++kt) {
    const float* row = (kt < 4) ? (pr + kt * 32) : (gr + (kt - 4) * 32);
    v16bf a = make_afrag(row + half * 8, row + 16 + half * 8);
#pragma unroll
    for (int nt = 0; nt < 8; ++nt) {
      v16bf b = *(const v16bf*)(Wfrag + (((kt * 8 + nt) * 32 + lane) << 4));
      acc[nt] = wmma_bf16(a, b, acc[nt]);
    }
  }
#pragma unroll
  for (int nt = 0; nt < 8; ++nt) {
    int col = nt * 16 + (lane & 15);
    float bias = bt[col];
#pragma unroll
    for (int i = 0; i < 8; ++i) {
      int gm = base + i + half * 8;
      long long off = (long long)gm * EMB + col;
      xgout[off] = lrelu(acc[nt][i] + bias) + xg[off];
    }
  }
}

extern "C" void kernel_launch(void* const* d_in, const int* in_sizes, int n_in,
                              void* d_out, int out_size, void* d_ws, size_t ws_size,
                              hipStream_t stream)
{
  const float* x0  = (const float*)d_in[0];
  const float* ea  = (const float*)d_in[1];
  const float* Wm  = (const float*)d_in[2];
  const float* bm  = (const float*)d_in[3];
  const float* Wa  = (const float*)d_in[4];
  const float* ba  = (const float*)d_in[5];
  const float* Wg  = (const float*)d_in[6];
  const float* bg  = (const float*)d_in[7];
  const float* Wf  = (const float*)d_in[8];
  const float* bfe = (const float*)d_in[9];
  const float* Wt  = (const float*)d_in[10];
  const float* bt  = (const float*)d_in[11];
  const long long* eidx  = (const long long*)d_in[12];
  const long long* batch = (const long long*)d_in[13];
  const int N = in_sizes[0] / EMB;
  const int E = in_sizes[1] / EDGEF;
  (void)n_in; (void)out_size; (void)ws_size;

  char* ws = (char*)d_ws;
  size_t o = 0;
  auto salloc = [&](size_t bytes) { char* p = ws + o; o = (o + bytes + 255) & ~(size_t)255; return p; };
  bf16_t* wmF  = (bf16_t*)salloc((size_t)NSTEPS * 160 * EMB * sizeof(bf16_t));
  bf16_t* waF  = (bf16_t*)salloc((size_t)NSTEPS * 384 * EMB * sizeof(bf16_t));
  bf16_t* wfF  = (bf16_t*)salloc((size_t)NSTEPS * 128 * EMB * sizeof(bf16_t));
  bf16_t* wtF  = (bf16_t*)salloc((size_t)NSTEPS * 256 * EMB * sizeof(bf16_t));
  float* agg    = (float*)salloc((size_t)N * EMB * 4);
  float* xA     = (float*)salloc((size_t)N * EMB * 4);
  float* xB     = (float*)salloc((size_t)N * EMB * 4);
  float* gate   = (float*)salloc((size_t)N * 4);
  float* evec   = (float*)salloc((size_t)N * 4);
  float* gmax   = (float*)salloc((size_t)GNUM * 4);
  float* denom  = (float*)salloc((size_t)GNUM * 4);
  float* pooled = (float*)salloc((size_t)GNUM * EMB * 4);
  float* xg0    = (float*)salloc((size_t)GNUM * EMB * 4);
  float* xg1    = (float*)salloc((size_t)GNUM * EMB * 4);

  // Repack weights to bf16 B-fragments (every call; deterministic)
  for (int s = 0; s < NSTEPS; ++s) {
    int kmn;
    kmn = 160 * EMB; pack_b_frags<<<(kmn + 255) / 256, 256, 0, stream>>>(Wm + (size_t)s * kmn, wmF + (size_t)s * kmn, 160);
    kmn = 384 * EMB; pack_b_frags<<<(kmn + 255) / 256, 256, 0, stream>>>(Wa + (size_t)s * kmn, waF + (size_t)s * kmn, 384);
    kmn = 128 * EMB; pack_b_frags<<<(kmn + 255) / 256, 256, 0, stream>>>(Wf + (size_t)s * kmn, wfF + (size_t)s * kmn, 128);
    kmn = 256 * EMB; pack_b_frags<<<(kmn + 255) / 256, 256, 0, stream>>>(Wt + (size_t)s * kmn, wtF + (size_t)s * kmn, 256);
  }

  const long long nxg = (long long)GNUM * EMB;
  fill_u32<<<(unsigned)((nxg + 255) / 256), 256, 0, stream>>>((unsigned*)xg0, 0u, nxg);

  float* xgCur = xg0;
  float* xgNxt = xg1;
  float* xOut  = (float*)d_out;
  float* xgOut = (float*)d_out + (size_t)N * EMB;

  const float* xCur = x0;
  for (int s = 0; s < NSTEPS; ++s) {
    float* xNext  = (s == 0) ? xA : (s == 1) ? xB : xOut;
    float* xgNext = (s == NSTEPS - 1) ? xgOut : xgNxt;

    const long long nAgg = (long long)N * EMB;
    fill_u32<<<(unsigned)((nAgg + 255) / 256), 256, 0, stream>>>((unsigned*)agg, 0xFF800000u, nAgg); // -inf
    edge_msg_scattermax<<<(E + 255) / 256, 256, 0, stream>>>(
        xCur, ea, eidx, wmF + (size_t)s * 160 * EMB, bm + s * EMB, agg, E);
    node_update<<<(N + 127) / 128, 256, 0, stream>>>(
        xCur, xgCur, agg, batch, waF + (size_t)s * 384 * EMB, ba + s * EMB, xNext, N);

    fill_u32<<<1, 256, 0, stream>>>((unsigned*)gmax, 0xFF800000u, GNUM);
    fill_u32<<<1, 256, 0, stream>>>((unsigned*)denom, 0u, GNUM);
    fill_u32<<<(unsigned)((nxg + 255) / 256), 256, 0, stream>>>((unsigned*)pooled, 0u, nxg);

    gate_max<<<(N + 255) / 256, 256, 0, stream>>>(xNext, Wg + s * EMB, bg + s, batch, gate, gmax, N);
    softmax_den<<<(N + 255) / 256, 256, 0, stream>>>(gate, gmax, batch, evec, denom, N);
    feat_pool<<<(N + 127) / 128, 256, 0, stream>>>(
        xNext, batch, evec, denom, wfF + (size_t)s * 128 * EMB, bfe + s * EMB, pooled, N);
    global_update<<<2, 256, 0, stream>>>(
        pooled, xgCur, wtF + (size_t)s * 256 * EMB, bt + s * EMB, xgNext);

    xCur = xNext;
    float* tmp = xgCur;
    xgCur = xgNext;
    if (s < NSTEPS - 1) xgNxt = tmp;
  }
}
